// DeformableTransformerLayer_274877907205
// MI455X (gfx1250) — compile-verified, matrix operand order checked by reading, not run on hardware
//
#include <hip/hip_runtime.h>
#include <hip/hip_bf16.h>
#include <math.h>

#define USE_ASYNC_STAGE 1   // stage LDS tiles with global_load_async_to_lds_b128

// ---------------------------------------------------------------------------
// Types for WMMA fragments (wave32, V_WMMA_F32_16X16X32_F16)
// ---------------------------------------------------------------------------
typedef _Float16 v16h __attribute__((ext_vector_type(16)));
typedef _Float16 v8h  __attribute__((ext_vector_type(8)));
typedef float    v8f  __attribute__((ext_vector_type(8)));

union Frag16 { v16h v; v8h h[2]; };

// Problem constants (from reference setup_inputs)
#define NB      2
#define NQ      8192
#define NC      256
#define NHEADS  8
#define NLVL    3
#define NPTS    4
#define DH      32
#define NV      6300        // 60*80 + 30*40 + 15*20
#define ROWS    (NB*NQ)     // 16384
#define VROWS   (NB*NV)     // 12600

// Epilogue ops for the generic GEMM
#define EPI_BIAS_F32  0   // out_f32[row*N+col] = acc + bias[col]
#define EPI_PERMV     1   // head-permuted value projection, f32
#define EPI_ATTNRES   2   // q = acc + bias + qn + query   (f32)
#define EPI_GELU_F16  3   // out_f16 = gelu(acc + bias)
#define EPI_FINAL     4   // d_out = qres + acc + bias     (f32)

static constexpr int LS = 40;   // LDS row stride in halves (32 + 8 pad, 16B-aligned)

// ---------------------------------------------------------------------------
// CDNA5 async global->LDS staging (ASYNCcnt-tracked), per CDNA5 ISA.
// GV mode: dest = LDS byte address in VGPR, source = 64-bit addr VGPR pair.
// ---------------------------------------------------------------------------
__device__ __forceinline__ void async_ld_b128(unsigned lds_off, const void* gptr)
{
    asm volatile("global_load_async_to_lds_b128 %0, %1, off"
                 :: "v"(lds_off), "v"(gptr) : "memory");
}
__device__ __forceinline__ void wait_async0()
{
    asm volatile("s_wait_asynccnt 0x0" ::: "memory");
}

// Stage a 128-row x 32-half tile (row stride K in memory) into LDS.
// 512 chunks of 8 halves (16B); 2 chunks per thread. OOB rows -> zero-fill.
__device__ __forceinline__ void stage_tile(const _Float16* __restrict__ src,
                                           int rowBase, int rowMax, int K, int k0,
                                           _Float16* lds, int tid)
{
    #pragma unroll
    for (int c = 0; c < 2; ++c) {
        int chunk = (tid << 1) | c;        // 0..511
        int row   = chunk >> 2;            // 0..127
        int off   = (chunk & 3) << 3;      // 0,8,16,24
        int gr    = rowBase + row;
        _Float16* dst = lds + row * LS + off;
        if (gr < rowMax) {
#if USE_ASYNC_STAGE
            async_ld_b128((unsigned)(size_t)dst, src + (size_t)gr * K + (k0 + off));
#else
            *(v8h*)dst = *(const v8h*)(src + (size_t)gr * K + (k0 + off));
#endif
        } else {
            v8h z = {};
            *(v8h*)dst = z;
        }
    }
}

// ---------------------------------------------------------------------------
// Generic wave32 WMMA GEMM:  C[M,N] = A[M,K](f16) * Bt[N,K](f16)^T (+ epilogue)
// B is pre-transposed on the host side of the pipeline (weights, done once),
// so both A and B tiles stage as contiguous 16B chunks and both fragment
// types are two 16-byte LDS reads per lane (documented 16-bit A/B layouts).
// Block: 256 threads = 8 waves. Block tile 128x128, K-step 32, double-buffered
// LDS with async staging overlapped against the WMMA burst; one barrier/iter.
// Wave tile 32(M) x 64(N) = 2x4 fragments -> 8 WMMA per K-step per wave.
// ---------------------------------------------------------------------------
template<int OP>
__global__ __launch_bounds__(256)
void gemm_wmma(const _Float16* __restrict__ A, const _Float16* __restrict__ Bt,
               const float* __restrict__ bias, const float* __restrict__ aux0,
               const float* __restrict__ aux1, void* __restrict__ out,
               int M, int N, int K)
{
    __shared__ __align__(16) _Float16 ldsA[2][128 * LS];
    __shared__ __align__(16) _Float16 ldsB[2][128 * LS];

    const int tid    = threadIdx.x;
    const int lane   = tid & 31;
    const int waveId = tid >> 5;
    const int wm     = waveId & 3;     // 4 waves along M (32 rows each)
    const int wn     = waveId >> 2;    // 2 waves along N (64 cols each)
    const int mlane  = lane & 15;
    const int hlf    = lane >> 4;

    const int blockM = blockIdx.x * 128;
    const int blockN = blockIdx.y * 128;

    v8f cacc[2][4] = {};

    const int kTiles = K >> 5;

    // Prologue: stage tile 0 into buffer 0
    stage_tile(A,  blockM, M, K, 0, ldsA[0], tid);
    stage_tile(Bt, blockN, N, K, 0, ldsB[0], tid);
#if USE_ASYNC_STAGE
    wait_async0();
#endif
    __syncthreads();

    for (int kt = 0; kt < kTiles; ++kt) {
        const int p = kt & 1;

        // Stage next K-tile into the other buffer (overlapped with compute)
        if (kt + 1 < kTiles) {
            const int k1 = (kt + 1) << 5;
            if (kt + 2 < kTiles) {    // far prefetch (global_prefetch_b8)
                int pr = blockM + (tid >> 1);
                if (pr < M) __builtin_prefetch(A + (size_t)pr * K + ((kt + 2) << 5), 0, 3);
            }
            stage_tile(A,  blockM, M, K, k1, ldsA[p ^ 1], tid);
            stage_tile(Bt, blockN, N, K, k1, ldsB[p ^ 1], tid);
        }

        // Compute on current buffer
        const _Float16* lA = ldsA[p];
        const _Float16* lB = ldsB[p];
        Frag16 af[2], bf[4];
        #pragma unroll
        for (int i = 0; i < 2; ++i) {
            int r = wm * 32 + i * 16 + mlane;
            af[i].h[0] = *(const v8h*)(lA + r * LS + hlf * 8);
            af[i].h[1] = *(const v8h*)(lA + r * LS + 16 + hlf * 8);
        }
        #pragma unroll
        for (int j = 0; j < 4; ++j) {
            int cN = wn * 64 + j * 16 + mlane;
            bf[j].h[0] = *(const v8h*)(lB + cN * LS + hlf * 8);
            bf[j].h[1] = *(const v8h*)(lB + cN * LS + 16 + hlf * 8);
        }
        #pragma unroll
        for (int i = 0; i < 2; ++i)
            #pragma unroll
            for (int j = 0; j < 4; ++j)
                cacc[i][j] = __builtin_amdgcn_wmma_f32_16x16x32_f16(
                    false, af[i].v, false, bf[j].v,
                    (short)0, cacc[i][j], false, false);

#if USE_ASYNC_STAGE
        wait_async0();     // async writes to the other buffer are done
#endif
        __syncthreads();   // everyone done reading p / staging p^1
    }

    // Epilogue: C layout lane(L): col = L&15, row = r + 8*(L>>4)
    #pragma unroll
    for (int i = 0; i < 2; ++i) {
        #pragma unroll
        for (int j = 0; j < 4; ++j) {
            int col = blockN + wn * 64 + j * 16 + mlane;
            #pragma unroll
            for (int r = 0; r < 8; ++r) {
                int row = blockM + wm * 32 + i * 16 + hlf * 8 + r;
                if (row < M && col < N) {
                    float acc = cacc[i][j][r] + bias[col];
                    if constexpr (OP == EPI_BIAS_F32) {
                        ((float*)out)[(size_t)row * N + col] = acc;
                    } else if constexpr (OP == EPI_PERMV) {
                        int b  = row / NV;
                        int ii = row - b * NV;
                        int h  = col >> 5;
                        int d  = col & 31;
                        ((float*)out)[(((size_t)(b * NHEADS + h)) * NV + ii) * DH + d] = acc;
                    } else if constexpr (OP == EPI_ATTNRES) {
                        size_t idx = (size_t)row * NC + col;
                        ((float*)out)[idx] = acc + aux0[idx] + aux1[idx];
                    } else if constexpr (OP == EPI_GELU_F16) {
                        float g = 0.5f * acc * (1.0f + erff(acc * 0.70710678118654752f));
                        ((_Float16*)out)[(size_t)row * N + col] = (_Float16)g;
                    } else if constexpr (OP == EPI_FINAL) {
                        size_t idx = (size_t)row * NC + col;
                        ((float*)out)[idx] = aux0[idx] + acc;
                    }
                }
            }
        }
    }
}

// ---------------------------------------------------------------------------
// f32 -> f16 conversion (value tensor: stays row-major, it's a GEMM A matrix)
// ---------------------------------------------------------------------------
__global__ void f32_to_f16_kernel(const float* __restrict__ in,
                                  _Float16* __restrict__ out, int n)
{
    int i = blockIdx.x * blockDim.x + threadIdx.x;
    if (i < n) out[i] = (_Float16)in[i];
}

// f32 [K][N] weight -> f16 [N][K] transposed (GEMM B matrices, done once)
__global__ void convert_transpose_kernel(const float* __restrict__ in,
                                         _Float16* __restrict__ out, int K, int N)
{
    int i = blockIdx.x * blockDim.x + threadIdx.x;
    if (i < K * N) {
        int n = i / K;
        int k = i - n * K;
        out[i] = (_Float16)in[(size_t)k * N + n];
    }
}

// ---------------------------------------------------------------------------
// LayerNorm over C=256; one 256-thread block per row.
// ---------------------------------------------------------------------------
template<bool ADD_POS, bool WRITE_F32>
__global__ __launch_bounds__(256)
void layernorm_kernel(const float* __restrict__ x, const float* __restrict__ pos,
                      const float* __restrict__ g, const float* __restrict__ bb,
                      float* __restrict__ out_f32, _Float16* __restrict__ out_f16)
{
    __shared__ float s1[256];
    __shared__ float s2[256];
    const int row = blockIdx.x;
    const int t   = threadIdx.x;
    const size_t idx = (size_t)row * NC + t;

    float v = x[idx];
    s1[t] = v;
    s2[t] = v * v;
    __syncthreads();
    #pragma unroll
    for (int s = 128; s > 0; s >>= 1) {
        if (t < s) { s1[t] += s1[t + s]; s2[t] += s2[t + s]; }
        __syncthreads();
    }
    float mean = s1[0] * (1.0f / NC);
    float var  = s2[0] * (1.0f / NC) - mean * mean;
    float r    = rsqrtf(var + 1e-5f);
    float y    = (v - mean) * r * g[t] + bb[t];
    if constexpr (WRITE_F32) out_f32[idx] = y;
    float y2 = y;
    if constexpr (ADD_POS) y2 = y + pos[idx];
    out_f16[idx] = (_Float16)y2;
}

// ---------------------------------------------------------------------------
// Softmax over 12 contiguous logits per (b,q,h); in-place.
// ---------------------------------------------------------------------------
__global__ void softmax12_kernel(float* __restrict__ p, int total)
{
    int i = blockIdx.x * blockDim.x + threadIdx.x;
    if (i >= total) return;
    float* q = p + (size_t)i * 12;
    float m = q[0];
    #pragma unroll
    for (int j = 1; j < 12; ++j) m = fmaxf(m, q[j]);
    float e[12], s = 0.f;
    #pragma unroll
    for (int j = 0; j < 12; ++j) { e[j] = expf(q[j] - m); s += e[j]; }
    float rs = 1.0f / s;
    #pragma unroll
    for (int j = 0; j < 12; ++j) q[j] = e[j] * rs;
}

// ---------------------------------------------------------------------------
// Deformable bilinear sampling: one wave per (b,q,h), lane = channel d (0..31).
// ---------------------------------------------------------------------------
__device__ __forceinline__ float ds_corner(const float* __restrict__ vl,
                                           int H, int W, int xi, int yi, int lane)
{
    if (xi < 0 || xi >= W || yi < 0 || yi >= H) return 0.f;
    return vl[((size_t)(yi * W + xi)) * DH + lane];
}

__global__ __launch_bounds__(256)
void deform_sample_kernel(const float* __restrict__ vproj,  // [B][h][NV][32]
                          const float* __restrict__ offs,   // [B*Nq][192]
                          const float* __restrict__ aw,     // [B*Nq][96]
                          const float* __restrict__ ref,    // [B*Nq][3][2]
                          _Float16* __restrict__ samp)      // [B*Nq][256]
{
    const int HWs[NLVL][2] = {{60, 80}, {30, 40}, {15, 20}};
    const int starts[NLVL] = {0, 4800, 6000};

    const int wave = blockIdx.x * 8 + (threadIdx.x >> 5);   // [0, 131072)
    const int lane = threadIdx.x & 31;
    const int h    = wave & (NHEADS - 1);
    const int bq   = wave >> 3;                             // [0, 16384)
    const int b    = bq >> 13;                              // Nq = 8192

    const float* vb  = vproj + ((size_t)(b * NHEADS + h)) * NV * DH;
    const float* of  = offs + (size_t)bq * (NHEADS * NLVL * NPTS * 2) + h * (NLVL * NPTS * 2);
    const float* awp = aw   + (size_t)bq * (NHEADS * NLVL * NPTS)     + h * (NLVL * NPTS);
    const float* rp  = ref  + (size_t)bq * (NLVL * 2);

    float acc = 0.f;
    #pragma unroll
    for (int l = 0; l < NLVL; ++l) {
        const int H = HWs[l][0], W = HWs[l][1];
        const float rx = rp[l * 2 + 0] * (float)W;
        const float ry = rp[l * 2 + 1] * (float)H;
        const float* vl = vb + (size_t)starts[l] * DH;
        #pragma unroll
        for (int p = 0; p < NPTS; ++p) {
            const float a  = awp[l * NPTS + p];
            const float gx = rx + of[(l * NPTS + p) * 2 + 0] - 0.5f;
            const float gy = ry + of[(l * NPTS + p) * 2 + 1] - 0.5f;
            const float x0f = floorf(gx), y0f = floorf(gy);
            const int   x0  = (int)x0f,  y0  = (int)y0f;
            const float wx = gx - x0f,   wy = gy - y0f;
            const float c00 = ds_corner(vl, H, W, x0,     y0,     lane);
            const float c10 = ds_corner(vl, H, W, x0 + 1, y0,     lane);
            const float c01 = ds_corner(vl, H, W, x0,     y0 + 1, lane);
            const float c11 = ds_corner(vl, H, W, x0 + 1, y0 + 1, lane);
            acc += a * (c00 * (1.f - wx) * (1.f - wy) + c10 * wx * (1.f - wy)
                      + c01 * (1.f - wx) * wy         + c11 * wx * wy);
        }
    }
    samp[(size_t)bq * NC + h * DH + lane] = (_Float16)acc;
}

// ---------------------------------------------------------------------------
// Host launcher
// ---------------------------------------------------------------------------
extern "C" void kernel_launch(void* const* d_in, const int* in_sizes, int n_in,
                              void* d_out, int out_size, void* d_ws, size_t ws_size,
                              hipStream_t stream)
{
    (void)in_sizes; (void)n_in; (void)out_size; (void)ws_size;

    const float* query     = (const float*)d_in[0];
    const float* value     = (const float*)d_in[1];
    const float* query_pos = (const float*)d_in[2];
    const float* ref_pts   = (const float*)d_in[3];
    // d_in[4] spatial_shapes, d_in[5] level_start_index: fixed constants
    const float* g1  = (const float*)d_in[6];
    const float* b1  = (const float*)d_in[7];
    const float* Wo  = (const float*)d_in[8];
    const float* bo  = (const float*)d_in[9];
    const float* Wa  = (const float*)d_in[10];
    const float* ba  = (const float*)d_in[11];
    const float* Wv  = (const float*)d_in[12];
    const float* bv  = (const float*)d_in[13];
    const float* Wp  = (const float*)d_in[14];
    const float* bp  = (const float*)d_in[15];
    const float* g2  = (const float*)d_in[16];
    const float* b2  = (const float*)d_in[17];
    const float* Wf1 = (const float*)d_in[18];
    const float* bf1 = (const float*)d_in[19];
    const float* Wf2 = (const float*)d_in[20];
    const float* bf2 = (const float*)d_in[21];

    // Workspace carving (256B aligned)
    char*  ws  = (char*)d_ws;
    size_t off = 0;
    auto alloc = [&](size_t bytes) -> void* {
        void* p = ws + off;
        off += (bytes + 255) & ~(size_t)255;
        return p;
    };

    float*     qn     = (float*)    alloc((size_t)ROWS * NC * 4);
    _Float16*  qa16   = (_Float16*) alloc((size_t)ROWS * NC * 2);
    _Float16*  Wo16   = (_Float16*) alloc((size_t)NC * 192 * 2);      // [192][256]
    _Float16*  Wa16   = (_Float16*) alloc((size_t)NC * 96 * 2);       // [96][256]
    _Float16*  Wv16   = (_Float16*) alloc((size_t)NC * NC * 2);       // [256][256]
    _Float16*  Wp16   = (_Float16*) alloc((size_t)NC * NC * 2);
    _Float16*  Wf116  = (_Float16*) alloc((size_t)NC * (4 * NC) * 2); // [1024][256]
    _Float16*  Wf216  = (_Float16*) alloc((size_t)(4 * NC) * NC * 2); // [256][1024]
    float*     offsb  = (float*)    alloc((size_t)ROWS * 192 * 4);
    float*     awb    = (float*)    alloc((size_t)ROWS * 96 * 4);
    _Float16*  val16  = (_Float16*) alloc((size_t)VROWS * NC * 2);
    float*     vproj  = (float*)    alloc((size_t)NB * NHEADS * NV * DH * 4);
    _Float16*  samp16 = (_Float16*) alloc((size_t)ROWS * NC * 2);
    float*     qbuf   = (float*)    alloc((size_t)ROWS * NC * 4);
    _Float16*  q216   = (_Float16*) alloc((size_t)ROWS * NC * 2);
    _Float16*  mid16  = (_Float16*) alloc((size_t)ROWS * (4 * NC) * 2);

    // 1) weight conversions (transposed to [N][K]) + value conversion
    convert_transpose_kernel<<<(NC * 192 + 255) / 256, 256, 0, stream>>>(Wo,  Wo16,  NC, 192);
    convert_transpose_kernel<<<(NC * 96 + 255) / 256, 256, 0, stream>>>(Wa,  Wa16,  NC, 96);
    convert_transpose_kernel<<<(NC * NC + 255) / 256, 256, 0, stream>>>(Wv,  Wv16,  NC, NC);
    convert_transpose_kernel<<<(NC * NC + 255) / 256, 256, 0, stream>>>(Wp,  Wp16,  NC, NC);
    convert_transpose_kernel<<<(NC * 4 * NC + 255) / 256, 256, 0, stream>>>(Wf1, Wf116, NC, 4 * NC);
    convert_transpose_kernel<<<(NC * 4 * NC + 255) / 256, 256, 0, stream>>>(Wf2, Wf216, 4 * NC, NC);
    f32_to_f16_kernel<<<(VROWS * NC + 255) / 256, 256, 0, stream>>>(value, val16, VROWS * NC);

    // 2) norm1 + positional embedding
    layernorm_kernel<true, true><<<ROWS, 256, 0, stream>>>(query, query_pos, g1, b1, qn, qa16);

    // 3) sampling offsets:  offs = qa @ Wo + bo   [16384 x 192]
    gemm_wmma<EPI_BIAS_F32><<<dim3(ROWS / 128, 2), 256, 0, stream>>>(
        qa16, Wo16, bo, nullptr, nullptr, offsb, ROWS, 192, NC);

    // 4) attention logits:  qa @ Wa + ba  [16384 x 96], then softmax over 12
    gemm_wmma<EPI_BIAS_F32><<<dim3(ROWS / 128, 1), 256, 0, stream>>>(
        qa16, Wa16, ba, nullptr, nullptr, awb, ROWS, 96, NC);
    softmax12_kernel<<<(ROWS * NHEADS) / 256, 256, 0, stream>>>(awb, ROWS * NHEADS);

    // 5) value projection, head-permuted out: v[b][h][i][d]
    gemm_wmma<EPI_PERMV><<<dim3((VROWS + 127) / 128, 2), 256, 0, stream>>>(
        val16, Wv16, bv, nullptr, nullptr, vproj, VROWS, NC, NC);

    // 6) deformable bilinear sampling + attention-weighted sum
    deform_sample_kernel<<<(ROWS * NHEADS) / 8, 256, 0, stream>>>(
        vproj, offsb, awb, ref_pts, samp16);

    // 7) q = samp @ Wp + bp + qn + query
    gemm_wmma<EPI_ATTNRES><<<dim3(ROWS / 128, 2), 256, 0, stream>>>(
        samp16, Wp16, bp, qn, query, qbuf, ROWS, NC, NC);

    // 8) norm2
    layernorm_kernel<false, false><<<ROWS, 256, 0, stream>>>(qbuf, nullptr, g2, b2, nullptr, q216);

    // 9) FFN up + exact GELU -> f16
    gemm_wmma<EPI_GELU_F16><<<dim3(ROWS / 128, 8), 256, 0, stream>>>(
        q216, Wf116, bf1, nullptr, nullptr, mid16, ROWS, 4 * NC, NC);

    // 10) FFN down + final residual -> d_out (f32)
    gemm_wmma<EPI_FINAL><<<dim3(ROWS / 128, 2), 256, 0, stream>>>(
        mid16, Wf216, bf2, qbuf, nullptr, d_out, ROWS, NC, 4 * NC);
}
